// MPNNLayer_62088047231703
// MI455X (gfx1250) — matrix operand order repaired and checked by reading, not compiled, not run on hardware
//
#include <hip/hip_runtime.h>
#include <hip/hip_bf16.h>
#include <math.h>

typedef __attribute__((ext_vector_type(16))) _Float16 v16h;
typedef __attribute__((ext_vector_type(8)))  _Float16 v8h;
typedef __attribute__((ext_vector_type(8)))  float    v8f;

#define WAVES_PER_BLOCK 8
#define TPW 2   // M-tiles (of 16 rows) per wave

// ---------------------------------------------------------------------------
// helpers
// ---------------------------------------------------------------------------

__device__ __forceinline__ void wave_lds_fence() {
    __builtin_amdgcn_wave_barrier();
    __asm__ volatile("s_wait_dscnt 0" ::: "memory");
    __builtin_amdgcn_wave_barrier();
}

// Build a 16x32 f16 A/B fragment from two contiguous 8-half (16B) runs.
__device__ __forceinline__ v16h frag_ld(const _Float16* p0, const _Float16* p1) {
    v8h lo = *(const v8h*)p0;
    v8h hh = *(const v8h*)p1;
    return __builtin_shufflevector(lo, hh, 0,1,2,3,4,5,6,7,8,9,10,11,12,13,14,15);
}

// Same, but source data is f32 (convert on the fly).
__device__ __forceinline__ v16h frag_ld_f32(const float* p0, const float* p1) {
    v8h lo = __builtin_convertvector(*(const v8f*)p0, v8h);
    v8h hh = __builtin_convertvector(*(const v8f*)p1, v8h);
    return __builtin_shufflevector(lo, hh, 0,1,2,3,4,5,6,7,8,9,10,11,12,13,14,15);
}

__device__ __forceinline__ float silu(float v) {
    return v / (1.0f + __expf(-v));
}

__device__ __forceinline__ v8f wmma_f16(v16h a, v16h b, v8f c) {
    return __builtin_amdgcn_wmma_f32_16x16x32_f16(
        /*neg_a=*/false, a, /*neg_b=*/false, b,
        /*c_mod=*/(short)0, c, /*reuse_a=*/false, /*reuse_b=*/false);
}

// ---------------------------------------------------------------------------
// prep kernels
// ---------------------------------------------------------------------------

__global__ void k_zero_f32(float* __restrict__ p, long n) {
    long i = (long)blockIdx.x * blockDim.x + threadIdx.x;
    long stride = (long)gridDim.x * blockDim.x;
    for (; i < n; i += stride) p[i] = 0.0f;
}

__global__ void k_cvt_f16(const float* __restrict__ s, _Float16* __restrict__ d, long n) {
    long i = (long)blockIdx.x * blockDim.x + threadIdx.x;
    long stride = (long)gridDim.x * blockDim.x;
    for (; i < n; i += stride) d[i] = (_Float16)s[i];
}

// W is [K x 128] row-major f32; Wt is [128 x K] row-major f16 (column-major view)
__global__ void k_transpose_f16(const float* __restrict__ W, _Float16* __restrict__ Wt, int K) {
    int idx = blockIdx.x * blockDim.x + threadIdx.x;
    if (idx >= K * 128) return;
    int k = idx >> 7;
    int nn = idx & 127;
    Wt[(size_t)nn * K + k] = (_Float16)W[idx];
}

// ---------------------------------------------------------------------------
// message kernel: 2 x 16-edge tiles per wave, weights staged in LDS
//   state[16 x 257] @ W1 -> SiLU -> @ W2 -> SiLU -> atomic scatter into aggr
// LDS layout (halves): [W1l: 32768][W2l: 16384][hbuf: 8 waves * 2 tiles * 2048]
// ---------------------------------------------------------------------------

__global__ void __launch_bounds__(256)
k_message(const int* __restrict__ ei, const float* __restrict__ pos,
          const _Float16* __restrict__ xh,
          const _Float16* __restrict__ W1t, const _Float16* __restrict__ W2t,
          const float* __restrict__ b1, const float* __restrict__ b2,
          const float* __restrict__ w1d,   // W1 row 256 (dist weights), f32
          float* __restrict__ aggr, int E)
{
    extern __shared__ _Float16 smem[];
    _Float16* W1l = smem;            // 256x128 f16, col-major(n,k) : 32768 halves
    _Float16* W2l = smem + 32768;    // 128x128 f16                 : 16384 halves

    // cooperative weight staging (global -> LDS), 16B granules
    {
        int4* d1 = (int4*)W1l;  const int4* s1 = (const int4*)W1t;
        for (int i = threadIdx.x; i < 4096; i += 256) d1[i] = s1[i];
        int4* d2 = (int4*)W2l;  const int4* s2 = (const int4*)W2t;
        for (int i = threadIdx.x; i < 2048; i += 256) d2[i] = s2[i];
    }
    __syncthreads();

    const int wave = threadIdx.x >> 5;
    const int lane = threadIdx.x & 31;
    const int m    = lane & 15;      // A-row within tile == C/D column
    const int hi   = lane >> 4;
    const int n    = m;
    const int ntiles = (E + 15) >> 4;
    const int base = (blockIdx.x * WAVES_PER_BLOCK + wave) * TPW;
    if (base >= ntiles) return;

    int  tiles[TPW]; bool tvalid[TPW];
    int  sN[TPW], rN[TPW];
    float dM[TPW][8]; int recM[TPW][8];

    #pragma unroll
    for (int p = 0; p < TPW; ++p) {
        int tile = base + p;
        tvalid[p] = (tile < ntiles);
        if (!tvalid[p]) tile = ntiles - 1;
        tiles[p] = tile;
        int e = tile * 16 + m;
        if (e >= E) e = E - 1;
        int s = ei[e];
        int r = ei[E + e];
        sN[p] = s; rN[p] = r;
        float dx = pos[3*s+0] - pos[3*r+0];
        float dy = pos[3*s+1] - pos[3*r+1];
        float dz = pos[3*s+2] - pos[3*r+2];
        float dist = sqrtf(dx*dx + dy*dy + dz*dz);
        #pragma unroll
        for (int i = 0; i < 8; ++i) {
            dM[p][i]   = __shfl(dist, i + 8*hi, 32);
            recM[p][i] = __shfl(r,    i + 8*hi, 32);
        }
    }

    // layer-1 accumulators: bias + dist * W1[256,:]
    v8f acc[TPW][8];
    #pragma unroll
    for (int t = 0; t < 8; ++t) {
        float bv = b1[t*16 + n];
        float wd = w1d[t*16 + n];
        #pragma unroll
        for (int p = 0; p < TPW; ++p)
            #pragma unroll
            for (int i = 0; i < 8; ++i)
                acc[p][t][i] = bv + dM[p][i] * wd;
    }

    // layer 1: K = 256 over x_s | x_r, B fragments from LDS, reused across TPW tiles
    #pragma unroll
    for (int c = 0; c < 8; ++c) {
        v16h A[TPW];
        #pragma unroll
        for (int p = 0; p < TPW; ++p) {
            const _Float16* row = (c < 4) ? (xh + (size_t)sN[p] * 128 + c*32)
                                          : (xh + (size_t)rN[p] * 128 + (c-4)*32);
            A[p] = frag_ld(row + hi*8, row + 16 + hi*8);
        }
        #pragma unroll
        for (int t = 0; t < 8; ++t) {
            const _Float16* bb = W1l + (t*16 + n) * 256 + c*32 + hi*8;
            v16h B = frag_ld(bb, bb + 16);
            #pragma unroll
            for (int p = 0; p < TPW; ++p)
                acc[p][t] = wmma_f16(A[p], B, acc[p][t]);
        }
    }

    // SiLU + transpose (D layout -> row-major) into per-wave LDS buffers
    _Float16* hb0 = smem + 49152 + (size_t)(wave * TPW) * 2048;
    #pragma unroll
    for (int p = 0; p < TPW; ++p) {
        _Float16* hb = hb0 + p * 2048;
        #pragma unroll
        for (int t = 0; t < 8; ++t)
            #pragma unroll
            for (int i = 0; i < 8; ++i)
                hb[(i + 8*hi) * 128 + t*16 + n] = (_Float16)silu(acc[p][t][i]);
    }
    wave_lds_fence();

    v8f acc2[TPW][8];
    #pragma unroll
    for (int t = 0; t < 8; ++t) {
        float bv = b2[t*16 + n];
        #pragma unroll
        for (int p = 0; p < TPW; ++p)
            #pragma unroll
            for (int i = 0; i < 8; ++i)
                acc2[p][t][i] = bv;
    }

    // layer 2: K = 128
    #pragma unroll
    for (int c = 0; c < 4; ++c) {
        v16h A[TPW];
        #pragma unroll
        for (int p = 0; p < TPW; ++p) {
            const _Float16* ab = hb0 + p * 2048 + m*128 + c*32;
            A[p] = frag_ld(ab + hi*8, ab + 16 + hi*8);
        }
        #pragma unroll
        for (int t = 0; t < 8; ++t) {
            const _Float16* bb = W2l + (t*16 + n) * 128 + c*32 + hi*8;
            v16h B = frag_ld(bb, bb + 16);
            #pragma unroll
            for (int p = 0; p < TPW; ++p)
                acc2[p][t] = wmma_f16(A[p], B, acc2[p][t]);
        }
    }

    // SiLU + fp32 atomic scatter-add into aggr[rec]
    #pragma unroll
    for (int p = 0; p < TPW; ++p) {
        #pragma unroll
        for (int t = 0; t < 8; ++t)
            #pragma unroll
            for (int i = 0; i < 8; ++i) {
                int M = i + 8*hi;
                if (tvalid[p] && (tiles[p] * 16 + M < E)) {
                    float msg = silu(acc2[p][t][i]);
                    __hip_atomic_fetch_add(&aggr[(size_t)recM[p][i] * 128 + t*16 + n],
                                           msg, __ATOMIC_RELAXED, __HIP_MEMORY_SCOPE_AGENT);
                }
            }
    }
}

// ---------------------------------------------------------------------------
// update kernel: 2 x 16-node tiles per wave, weights staged in LDS
//   [x | aggr] @ U1 -> SiLU -> @ U2 + c2 -> out
// ---------------------------------------------------------------------------

__global__ void __launch_bounds__(256)
k_update(const _Float16* __restrict__ xh, const float* __restrict__ aggr,
         const _Float16* __restrict__ U1t, const _Float16* __restrict__ U2t,
         const float* __restrict__ c1, const float* __restrict__ c2,
         float* __restrict__ out, int N)
{
    extern __shared__ _Float16 smem[];
    _Float16* U1l = smem;            // 32768 halves
    _Float16* U2l = smem + 32768;    // 16384 halves

    {
        int4* d1 = (int4*)U1l;  const int4* s1 = (const int4*)U1t;
        for (int i = threadIdx.x; i < 4096; i += 256) d1[i] = s1[i];
        int4* d2 = (int4*)U2l;  const int4* s2 = (const int4*)U2t;
        for (int i = threadIdx.x; i < 2048; i += 256) d2[i] = s2[i];
    }
    __syncthreads();

    const int wave = threadIdx.x >> 5;
    const int lane = threadIdx.x & 31;
    const int m    = lane & 15;
    const int hi   = lane >> 4;
    const int n    = m;
    const int ntiles = (N + 15) >> 4;
    const int base = (blockIdx.x * WAVES_PER_BLOCK + wave) * TPW;
    if (base >= ntiles) return;

    int tiles[TPW]; bool tvalid[TPW]; int nodeP[TPW];
    #pragma unroll
    for (int p = 0; p < TPW; ++p) {
        int tile = base + p;
        tvalid[p] = (tile < ntiles);
        if (!tvalid[p]) tile = ntiles - 1;
        tiles[p] = tile;
        int node = tile * 16 + m;
        if (node >= N) node = N - 1;
        nodeP[p] = node;
    }

    v8f acc[TPW][8];
    #pragma unroll
    for (int t = 0; t < 8; ++t) {
        float bv = c1[t*16 + n];
        #pragma unroll
        for (int p = 0; p < TPW; ++p)
            #pragma unroll
            for (int i = 0; i < 8; ++i)
                acc[p][t][i] = bv;
    }

    #pragma unroll
    for (int c = 0; c < 8; ++c) {
        v16h A[TPW];
        #pragma unroll
        for (int p = 0; p < TPW; ++p) {
            if (c < 4) {
                const _Float16* ab = xh + (size_t)nodeP[p] * 128 + c*32;
                A[p] = frag_ld(ab + hi*8, ab + 16 + hi*8);
            } else {
                const float* af = aggr + (size_t)nodeP[p] * 128 + (c-4)*32;
                A[p] = frag_ld_f32(af + hi*8, af + 16 + hi*8);
            }
        }
        #pragma unroll
        for (int t = 0; t < 8; ++t) {
            const _Float16* bb = U1l + (t*16 + n) * 256 + c*32 + hi*8;
            v16h B = frag_ld(bb, bb + 16);
            #pragma unroll
            for (int p = 0; p < TPW; ++p)
                acc[p][t] = wmma_f16(A[p], B, acc[p][t]);
        }
    }

    _Float16* ub0 = smem + 49152 + (size_t)(wave * TPW) * 2048;
    #pragma unroll
    for (int p = 0; p < TPW; ++p) {
        _Float16* ub = ub0 + p * 2048;
        #pragma unroll
        for (int t = 0; t < 8; ++t)
            #pragma unroll
            for (int i = 0; i < 8; ++i)
                ub[(i + 8*hi) * 128 + t*16 + n] = (_Float16)silu(acc[p][t][i]);
    }
    wave_lds_fence();

    v8f acc2[TPW][8];
    #pragma unroll
    for (int t = 0; t < 8; ++t) {
        float bv = c2[t*16 + n];
        #pragma unroll
        for (int p = 0; p < TPW; ++p)
            #pragma unroll
            for (int i = 0; i < 8; ++i)
                acc2[p][t][i] = bv;
    }

    #pragma unroll
    for (int c = 0; c < 4; ++c) {
        v16h A[TPW];
        #pragma unroll
        for (int p = 0; p < TPW; ++p) {
            const _Float16* ab = ub0 + p * 2048 + m*128 + c*32;
            A[p] = frag_ld(ab + hi*8, ab + 16 + hi*8);
        }
        #pragma unroll
        for (int t = 0; t < 8; ++t) {
            const _Float16* bb = U2l + (t*16 + n) * 128 + c*32 + hi*8;
            v16h B = frag_ld(bb, bb + 16);
            #pragma unroll
            for (int p = 0; p < TPW; ++p)
                acc2[p][t] = wmma_f16(A[p], B, acc2[p][t]);
        }
    }

    // final layer: no activation, direct store
    #pragma unroll
    for (int p = 0; p < TPW; ++p) {
        #pragma unroll
        for (int t = 0; t < 8; ++t)
            #pragma unroll
            for (int i = 0; i < 8; ++i) {
                int M = i + 8*hi;
                int nd = tiles[p] * 16 + M;
                if (tvalid[p] && nd < N)
                    out[(size_t)nd * 128 + t*16 + n] = acc2[p][t][i];
            }
    }
}

// ---------------------------------------------------------------------------
// launch
// ---------------------------------------------------------------------------

extern "C" void kernel_launch(void* const* d_in, const int* in_sizes, int n_in,
                              void* d_out, int out_size, void* d_ws, size_t ws_size,
                              hipStream_t stream) {
    const float* x   = (const float*)d_in[0];
    const float* pos = (const float*)d_in[1];
    const int*   ei  = (const int*)d_in[2];
    const float* W1  = (const float*)d_in[3];
    const float* b1  = (const float*)d_in[4];
    const float* W2  = (const float*)d_in[5];
    const float* b2  = (const float*)d_in[6];
    const float* U1  = (const float*)d_in[7];
    const float* c1  = (const float*)d_in[8];
    const float* U2  = (const float*)d_in[9];
    const float* c2  = (const float*)d_in[10];

    const int N = in_sizes[0] / 128;
    const int E = in_sizes[2] / 2;
    const long NH = (long)N * 128;

    // workspace layout (all region sizes are multiples of 256 bytes)
    char* ws = (char*)d_ws;
    _Float16* xh  = (_Float16*)ws;                 size_t off = (size_t)NH * 2;
    _Float16* W1t = (_Float16*)(ws + off);         off += 256 * 128 * 2;
    _Float16* W2t = (_Float16*)(ws + off);         off += 128 * 128 * 2;
    _Float16* U1t = (_Float16*)(ws + off);         off += 256 * 128 * 2;
    _Float16* U2t = (_Float16*)(ws + off);         off += 128 * 128 * 2;
    float*    aggr = (float*)(ws + off);

    // dynamic LDS: 96KB weights + 8 waves * TPW tiles * 4KB transpose buffers
    const size_t smem_bytes = (49152 + WAVES_PER_BLOCK * TPW * 2048) * sizeof(_Float16);

    // --- prep ---
    {
        int blk = 256;
        int g = (int)((NH + blk - 1) / blk);
        k_zero_f32<<<g, blk, 0, stream>>>(aggr, NH);
        k_cvt_f16 <<<g, blk, 0, stream>>>(x, xh, NH);
        k_transpose_f16<<<(256*128 + 255)/256, 256, 0, stream>>>(W1, W1t, 256);
        k_transpose_f16<<<(128*128 + 255)/256, 256, 0, stream>>>(W2, W2t, 128);
        k_transpose_f16<<<(256*128 + 255)/256, 256, 0, stream>>>(U1, U1t, 256);
        k_transpose_f16<<<(128*128 + 255)/256, 256, 0, stream>>>(U2, U2t, 128);
    }

    // --- message + scatter ---
    {
        int ntiles = (E + 15) / 16;
        int npairs = (ntiles + TPW - 1) / TPW;
        int grid = (npairs + WAVES_PER_BLOCK - 1) / WAVES_PER_BLOCK;
        const float* w1d = W1 + (size_t)256 * 128;   // dist row of W1, f32
        k_message<<<grid, 32 * WAVES_PER_BLOCK, smem_bytes, stream>>>(
            ei, pos, xh, W1t, W2t, b1, b2, w1d, aggr, E);
    }

    // --- update ---
    {
        int ntiles = (N + 15) / 16;
        int npairs = (ntiles + TPW - 1) / TPW;
        int grid = (npairs + WAVES_PER_BLOCK - 1) / WAVES_PER_BLOCK;
        k_update<<<grid, 32 * WAVES_PER_BLOCK, smem_bytes, stream>>>(
            xh, aggr, U1t, U2t, c1, c2, (float*)d_out, N);
    }
}
